// WindowedMamba_24567212933444
// MI455X (gfx1250) — matrix operand order repaired and checked by reading, not compiled, not run on hardware
//
#include <hip/hip_runtime.h>
#include <hip/hip_bf16.h>
#include <math.h>

typedef __bf16 bf16;
typedef __attribute__((ext_vector_type(16))) __bf16 v16bf;
typedef __attribute__((ext_vector_type(8)))  __bf16 v8bf;
typedef __attribute__((ext_vector_type(8)))  float  v8f;

#define MTOK 16384   // BN * L = 256 * 64 token rows

// ---------- helpers ----------
__device__ __forceinline__ bf16 f2bf(float f) {
  unsigned u = __builtin_bit_cast(unsigned, f);
  unsigned r = (u + 0x7FFFu + ((u >> 16) & 1u)) >> 16;   // RNE
  return __builtin_bit_cast(bf16, (unsigned short)r);
}
__device__ __forceinline__ float gelu_f(float x) {
  return 0.5f * x * (1.0f + erff(x * 0.70710678118654752f));
}
__device__ __forceinline__ float silu_f(float x) {
  return x / (1.0f + __expf(-x));
}
__device__ __forceinline__ v16bf cat8(v8bf lo, v8bf hi) {
  return __builtin_shufflevector(lo, hi, 0,1,2,3,4,5,6,7,8,9,10,11,12,13,14,15);
}
__device__ __forceinline__ v8f wmma_bf16(v16bf a, v16bf b, v8f c) {
  return __builtin_amdgcn_wmma_f32_16x16x32_bf16(false, a, false, b, (short)0, c, false, false);
}

// CDNA5 async global->LDS copy (GLOBAL_LOAD_ASYNC_TO_LDS_B128, ASYNCcnt).
// lds_off: low 32 bits of the flat shared address == LDS byte offset
// (ISA 10.2: the LDS offset is the low address dword; aperture lives in addr[63:32]).
__device__ __forceinline__ void async_copy_b128(unsigned lds_off, const void* gptr) {
  asm volatile("global_load_async_to_lds_b128 %0, %1, off"
               :: "v"(lds_off), "v"((unsigned long long)(size_t)gptr)
               : "memory");
}
__device__ __forceinline__ void async_wait0() {
  asm volatile("s_wait_asynccnt 0x0" ::: "memory");
}

// ---------- generic bf16 WMMA GEMM: C = A(MxK) * B(KxN) ----------
// TRANSB: B stored as (N x K) row-major (weight transpose access)
// EPI: 0 none, 1 +bias, 2 +bias+GELU
// OMODE: 0 f32 out, 1 bf16 out, 2 both, 3 f32 scatter into (b, c_off+col, s)
template<int TRANSB, int EPI, int OMODE>
__global__ __launch_bounds__(256) void gemm_bf16_wmma(
    const bf16* __restrict__ A, const bf16* __restrict__ B,
    const float* __restrict__ bias, float* __restrict__ Cf,
    bf16* __restrict__ Cb, int M, int N, int K, int c_off)
{
  __shared__ alignas(16) bf16 sA[128 * 32];
  __shared__ alignas(16) bf16 sB[32 * 64];
  const int tid   = threadIdx.x;
  const int lane  = tid & 31;
  const int wave  = tid >> 5;
  const int warpM = wave & 3;       // 4 waves along M
  const int warpN = wave >> 2;      // 2 waves along N
  const int mBase = blockIdx.x * 128;
  const int nBase = blockIdx.y * 64;

  v8f acc[2][2];
#pragma unroll
  for (int i = 0; i < 2; ++i)
#pragma unroll
    for (int j = 0; j < 2; ++j)
#pragma unroll
      for (int v = 0; v < 8; ++v) acc[i][j][v] = 0.0f;

  const unsigned ldsA0 = (unsigned)(size_t)(void*)sA;
  const unsigned ldsB0 = (unsigned)(size_t)(void*)sB;
  unsigned* sB32 = (unsigned*)sB;

  for (int k0 = 0; k0 < K; k0 += 32) {
    // stage A tile 128x32 via async global->LDS b128 copies (16B chunks)
#pragma unroll
    for (int j = 0; j < 2; ++j) {
      int i = tid + j * 256;                   // < 512 chunks: row=i>>2, sub=i&3
      int row = i >> 2, sub = i & 3;
      async_copy_b128(ldsA0 + i * 16,
                      (const void*)(A + (size_t)(mBase + row) * K + k0 + sub * 8));
    }
    // stage B tile 32x64
    if (TRANSB) {
#pragma unroll
      for (int j = 0; j < 4; ++j) {
        int idx = tid + j * 256;               // < 1024 pairs
        int kr = idx >> 5, nc = idx & 31;
        int n0 = nBase + nc * 2;
        unsigned short lo = 0, hi = 0;
        if (n0 < N)     lo = __builtin_bit_cast(unsigned short, B[(size_t)n0 * K + k0 + kr]);
        if (n0 + 1 < N) hi = __builtin_bit_cast(unsigned short, B[(size_t)(n0 + 1) * K + k0 + kr]);
        sB32[kr * 32 + nc] = (unsigned)lo | ((unsigned)hi << 16);
      }
    } else {
      int row = tid >> 3, sub = tid & 7;       // 256 chunks: row<32, sub<8
      int ncol = nBase + sub * 8;
      if (ncol < N)                            // skipped chunks feed only cols>=N (never stored)
        async_copy_b128(ldsB0 + tid * 16,
                        (const void*)(B + (size_t)(k0 + row) * N + ncol));
    }
    // prefetch next tiles into cache while this tile computes (global_prefetch_b8)
    if (k0 + 32 < K) {
      {
        int row = tid >> 4;
        __builtin_prefetch((const void*)(A + (size_t)(mBase + row) * K + k0 + 32), 0, 1);
      }
      if (!TRANSB) {
        int kr = tid >> 5, nc = tid & 31;
        int ncol = nBase + nc * 2;
        if (ncol + 1 < N)
          __builtin_prefetch((const void*)(B + (size_t)(k0 + 32 + kr) * N + ncol), 0, 1);
      } else {
        int n0 = nBase + ((tid & 31) * 2);
        if (n0 < N)
          __builtin_prefetch((const void*)(B + (size_t)n0 * K + k0 + 32 + (tid >> 5)), 0, 1);
      }
    }
    async_wait0();          // drain ASYNCcnt before the barrier
    __syncthreads();

#pragma unroll
    for (int mi = 0; mi < 2; ++mi) {
      const int half = lane >> 4;
      const int m = warpM * 32 + mi * 16 + (lane & 15);
      // A frag per ISA 16-bit 16x32 layout: e0..7 -> K=half*8+0..7, e8..15 -> K=16+half*8+0..7
      v8bf alo = *(const v8bf*)(sA + m * 32 + half * 8);
      v8bf ahi = *(const v8bf*)(sA + m * 32 + 16 + half * 8);
      v16bf a = cat8(alo, ahi);
#pragma unroll
      for (int ni = 0; ni < 2; ++ni) {
        // B frag: lane -> N, elements -> K in the same order as A
        const int nn = warpN * 32 + ni * 16 + (lane & 15);
        v16bf b;
#pragma unroll
        for (int s = 0; s < 16; ++s) b[s] = sB[(half * 8 + (s & 7) + ((s >> 3) << 4)) * 64 + nn];
        acc[mi][ni] = wmma_bf16(a, b, acc[mi][ni]);
      }
    }
    __syncthreads();
  }

  // epilogue: C/D f32 layout: lane->N, vgpr v -> M = v + 8*(lane>>4)
#pragma unroll
  for (int mi = 0; mi < 2; ++mi)
#pragma unroll
    for (int ni = 0; ni < 2; ++ni) {
      const int col = nBase + warpN * 32 + ni * 16 + (lane & 15);
      float bv = 0.0f;
      if (EPI >= 1 && col < N) bv = bias[col];
#pragma unroll
      for (int v = 0; v < 8; ++v) {
        const int row = mBase + warpM * 32 + mi * 16 + v + ((lane >> 4) << 3);
        float val = acc[mi][ni][v];
        if (EPI >= 1) val += bv;
        if (EPI == 2) val = gelu_f(val);
        if (col < N) {
          if (OMODE == 0)      Cf[(size_t)row * N + col] = val;
          else if (OMODE == 1) Cb[(size_t)row * N + col] = f2bf(val);
          else if (OMODE == 2) { Cf[(size_t)row * N + col] = val;
                                 Cb[(size_t)row * N + col] = f2bf(val); }
          else { // scatter into out (4, 512, 4096)
            const int b_ = row >> 12, s_ = row & 4095;
            Cf[((size_t)(b_ * 512 + c_off + col)) * 4096 + s_] = val;
          }
        }
      }
    }
}

// ---------- patch extract + f32->bf16 ----------
__global__ __launch_bounds__(256) void patchify_kernel(
    const float* __restrict__ x, const float* __restrict__ q,
    bf16* __restrict__ xm, bf16* __restrict__ xsa, bf16* __restrict__ qp)
{
  size_t idx = (size_t)blockIdx.x * 256 + threadIdx.x;   // t*256 + c
  int c = (int)(idx & 255);
  int t = (int)(idx >> 8);
  int b = t >> 12;
  int s = t & 4095;                                      // n*64 + w
  xm[idx]  = f2bf(x[((size_t)(b * 512 + c)) * 4096 + s]);
  xsa[idx] = f2bf(x[((size_t)(b * 512 + 256 + c)) * 4096 + s]);
  qp[idx]  = f2bf(q[((size_t)(b * 256 + c)) * 4096 + s]);
}

__global__ __launch_bounds__(256) void cvt_kernel(
    const float* __restrict__ src, bf16* __restrict__ dst, int n)
{
  int i = blockIdx.x * 256 + threadIdx.x;
  if (i < n) dst[i] = f2bf(src[i]);
}

// ---------- depthwise causal conv (width 4) + SiLU ----------
__global__ __launch_bounds__(256) void conv_silu_kernel(
    const float* __restrict__ xz, const float* __restrict__ cw,
    const float* __restrict__ cb, float* __restrict__ xcf, bf16* __restrict__ xcb)
{
  size_t idx = (size_t)blockIdx.x * 256 + threadIdx.x;   // t*512 + d
  int d = (int)(idx & 511);
  int t = (int)(idx >> 9);
  int w = t & 63;
  int t0 = t - w;
  float o = cb[d];
#pragma unroll
  for (int j = 0; j < 4; ++j) {
    int ww = w - 3 + j;
    if (ww >= 0) o += cw[d * 4 + j] * xz[(size_t)(t0 + ww) * 1024 + d];
  }
  float v = silu_f(o);
  xcf[idx] = v;
  xcb[idx] = f2bf(v);
}

// ---------- dt = softplus(dt_r @ dt_proj_w + b)  (K=16, VALU) ----------
__global__ __launch_bounds__(256) void dt_kernel(
    const float* __restrict__ dbl, const float* __restrict__ dtw,
    const float* __restrict__ dtb, float* __restrict__ dtf)
{
  size_t idx = (size_t)blockIdx.x * 256 + threadIdx.x;   // t*512 + d
  int d = (int)(idx & 511);
  int t = (int)(idx >> 9);
  float a = dtb[d];
#pragma unroll
  for (int r = 0; r < 16; ++r) a += dbl[(size_t)t * 48 + r] * dtw[r * 512 + d];
  dtf[idx] = (a > 20.0f) ? a : log1pf(__expf(a));
}

// ---------- bidirectional selective scan: one thread per (bn, d) ----------
__global__ __launch_bounds__(256) void scan_kernel(
    const float* __restrict__ u, const float* __restrict__ dtf,
    const float* __restrict__ dbl, const float* __restrict__ xz,
    const float* __restrict__ A_log, const float* __restrict__ Ab_log,
    const float* __restrict__ Dp, float* __restrict__ ytmp, bf16* __restrict__ yb)
{
  int idx = blockIdx.x * 256 + threadIdx.x;              // bn*512 + d, 131072 total
  int d  = idx & 511;
  int bn = idx >> 9;
  float Af[16], Ab[16], h[16];
#pragma unroll
  for (int n = 0; n < 16; ++n) {
    Af[n] = -__expf(A_log[d * 16 + n]);
    Ab[n] = -__expf(Ab_log[d * 16 + n]);
    h[n] = 0.0f;
  }
  const float Dd = Dp[d];
  for (int w = 0; w < 64; ++w) {
    int t = bn * 64 + w;
    float uv = u[(size_t)t * 512 + d];
    float dv = dtf[(size_t)t * 512 + d];
    float zv = xz[(size_t)t * 1024 + 512 + d];
    float y = 0.0f;
#pragma unroll
    for (int n = 0; n < 16; ++n) {
      h[n] = __expf(dv * Af[n]) * h[n] + dv * uv * dbl[(size_t)t * 48 + 16 + n];
      y += h[n] * dbl[(size_t)t * 48 + 32 + n];
    }
    ytmp[(size_t)t * 512 + d] = (y + uv * Dd) * silu_f(zv);
  }
#pragma unroll
  for (int n = 0; n < 16; ++n) h[n] = 0.0f;
  for (int w = 63; w >= 0; --w) {
    int t = bn * 64 + w;
    float uv = u[(size_t)t * 512 + d];
    float dv = dtf[(size_t)t * 512 + d];
    float zv = xz[(size_t)t * 1024 + 512 + d];
    float y = 0.0f;
#pragma unroll
    for (int n = 0; n < 16; ++n) {
      h[n] = __expf(dv * Ab[n]) * h[n] + dv * uv * dbl[(size_t)t * 48 + 16 + n];
      y += h[n] * dbl[(size_t)t * 48 + 32 + n];
    }
    float tot = ytmp[(size_t)t * 512 + d] + (y + uv * Dd) * silu_f(zv);
    yb[(size_t)t * 512 + d] = f2bf(tot);
  }
}

// ---------- windowed attention: one block per (window, head) ----------
// K staged TRANSPOSED in LDS (ld=72 keeps b128 frag loads 16B-aligned and
// breaks the transpose-store bank conflict); S = Q Kt /16 and ctx = P V via
// WMMA; f32 softmax; GELU fused into the ctx store.
#define KT_LD 72
__global__ __launch_bounds__(256) void attn_kernel(
    const bf16* __restrict__ Q, const bf16* __restrict__ K,
    const bf16* __restrict__ V, bf16* __restrict__ G)
{
  __shared__ alignas(16) bf16  sKT[256 * KT_LD];  // 36KB: K^T, [c][w]
  __shared__ alignas(16) float sS[64 * 64];       // 16KB
  __shared__ alignas(16) bf16  sP[64 * 64];       // 8KB
  const int tid = threadIdx.x, lane = tid & 31, wave = tid >> 5;
  const int bn = blockIdx.x >> 3, h = blockIdx.x & 7;
  const size_t rowBase = (size_t)bn * 64 * 2048 + (size_t)h * 256;

  // stage K window (64 x 256) transposed into LDS
  const unsigned* K32 = (const unsigned*)K;
#pragma unroll
  for (int j = 0; j < 32; ++j) {
    int idx = tid + j * 256;                   // pair index < 8192
    int w = idx >> 7, cp = idx & 127;
    unsigned pr = K32[(rowBase + (size_t)w * 2048) / 2 + cp];
    sKT[(size_t)(cp * 2) * KT_LD + w]     = __builtin_bit_cast(bf16, (unsigned short)(pr & 0xFFFFu));
    sKT[(size_t)(cp * 2 + 1) * KT_LD + w] = __builtin_bit_cast(bf16, (unsigned short)(pr >> 16));
  }
  __syncthreads();

  { // S = Q K^T: 16 tiles of 16x16, 2 per wave (shared A frag)
    const int mi  = wave >> 1;
    const int ni0 = (wave & 1) * 2;
    v8f acc0, acc1;
#pragma unroll
    for (int v = 0; v < 8; ++v) { acc0[v] = 0.0f; acc1[v] = 0.0f; }
    const int half = lane >> 4;
    const int mrow = mi * 16 + (lane & 15);
    const bf16* Qrow = Q + rowBase + (size_t)mrow * 2048;
    for (int k0 = 0; k0 < 256; k0 += 32) {
      v8bf alo = *(const v8bf*)(Qrow + k0 + half * 8);
      v8bf ahi = *(const v8bf*)(Qrow + k0 + 16 + half * 8);
      v16bf a = cat8(alo, ahi);
      v16bf b0, b1;
#pragma unroll
      for (int s = 0; s < 16; ++s) {
        const bf16* kt = sKT + (size_t)(k0 + half * 8 + (s & 7) + ((s >> 3) << 4)) * KT_LD + (lane & 15);
        b0[s] = kt[ni0 * 16];
        b1[s] = kt[(ni0 + 1) * 16];
      }
      acc0 = wmma_bf16(a, b0, acc0);
      acc1 = wmma_bf16(a, b1, acc1);
    }
    const float sc = 1.0f / 16.0f;                           // 1/sqrt(256)
#pragma unroll
    for (int v = 0; v < 8; ++v) {
      int r = mi * 16 + v + ((lane >> 4) << 3);
      sS[r * 64 + ni0 * 16 + (lane & 15)]       = acc0[v] * sc;
      sS[r * 64 + (ni0 + 1) * 16 + (lane & 15)] = acc1[v] * sc;
    }
  }
  __syncthreads();

  if (tid < 64) { // row softmax in f32, emit bf16 P
    float mx = -1e30f;
    for (int j = 0; j < 64; ++j) mx = fmaxf(mx, sS[tid * 64 + j]);
    float sum = 0.0f;
    for (int j = 0; j < 64; ++j) { float e = __expf(sS[tid * 64 + j] - mx); sS[tid * 64 + j] = e; sum += e; }
    float inv = 1.0f / sum;
    for (int j = 0; j < 64; ++j) sP[tid * 64 + j] = f2bf(sS[tid * 64 + j] * inv);
  }
  __syncthreads();

  { // ctx = P(64x64) @ V(64x256): 64 tiles, 8 per wave; GELU + store to G
    const int mi   = wave & 3;
    const int nset = wave >> 2;
    v8f acc[8];
#pragma unroll
    for (int nj = 0; nj < 8; ++nj)
#pragma unroll
      for (int v = 0; v < 8; ++v) acc[nj][v] = 0.0f;
    const int half = lane >> 4;
    const int mrow = mi * 16 + (lane & 15);
    for (int k0 = 0; k0 < 64; k0 += 32) {
      v8bf alo = *(const v8bf*)(sP + mrow * 64 + k0 + half * 8);
      v8bf ahi = *(const v8bf*)(sP + mrow * 64 + k0 + 16 + half * 8);
      v16bf a = cat8(alo, ahi);
#pragma unroll
      for (int nj = 0; nj < 8; ++nj) {
        const int n0 = (nset * 8 + nj) * 16 + (lane & 15);
        v16bf b;
#pragma unroll
        for (int s = 0; s < 16; ++s) b[s] = V[rowBase + (size_t)(k0 + half * 8 + (s & 7) + ((s >> 3) << 4)) * 2048 + n0];
        acc[nj] = wmma_bf16(a, b, acc[nj]);
      }
    }
#pragma unroll
    for (int nj = 0; nj < 8; ++nj) {
      const int n0 = (nset * 8 + nj) * 16 + (lane & 15);
#pragma unroll
      for (int v = 0; v < 8; ++v) {
        const int r = mi * 16 + v + ((lane >> 4) << 3);
        G[rowBase + (size_t)r * 2048 + n0] = f2bf(gelu_f(acc[nj][v]));
      }
    }
  }
}

// ---------- host ----------
extern "C" void kernel_launch(void* const* d_in, const int* in_sizes, int n_in,
                              void* d_out, int out_size, void* d_ws, size_t ws_size,
                              hipStream_t stream)
{
  (void)in_sizes; (void)n_in; (void)out_size; (void)ws_size;
  const float* x         = (const float*)d_in[0];
  const float* q         = (const float*)d_in[1];
  const float* in_proj_w = (const float*)d_in[2];
  const float* conv_w    = (const float*)d_in[3];
  const float* conv_b    = (const float*)d_in[4];
  const float* x_proj_w  = (const float*)d_in[5];
  const float* dt_proj_w = (const float*)d_in[6];
  const float* dt_proj_b = (const float*)d_in[7];
  const float* A_log     = (const float*)d_in[8];
  const float* A_b_log   = (const float*)d_in[9];
  const float* Dparam    = (const float*)d_in[10];
  const float* out_proj_w= (const float*)d_in[11];
  const float* wq        = (const float*)d_in[12];
  const float* wk        = (const float*)d_in[13];
  const float* wv        = (const float*)d_in[14];
  const float* proj_w    = (const float*)d_in[15];
  const float* proj_b    = (const float*)d_in[16];
  const float* fc1_w     = (const float*)d_in[17];
  const float* fc1_b     = (const float*)d_in[18];
  const float* fc2_w     = (const float*)d_in[19];
  const float* fc2_b     = (const float*)d_in[20];
  float* out = (float*)d_out;

  size_t off = 0;
  auto alloc = [&](size_t bytes) -> void* {
    void* p = (char*)d_ws + off;
    off += (bytes + 255) & ~(size_t)255;
    return p;
  };
  // bf16 activations / weights
  bf16* xm_b    = (bf16*)alloc((size_t)MTOK * 256 * 2);
  bf16* xsa_b   = (bf16*)alloc((size_t)MTOK * 256 * 2);
  bf16* qp_b    = (bf16*)alloc((size_t)MTOK * 256 * 2);
  bf16* w_in_b  = (bf16*)alloc(262144 * 2);
  bf16* w_xp_b  = (bf16*)alloc(24576 * 2);
  bf16* w_out_b = (bf16*)alloc(131072 * 2);
  bf16* w_fc1_b = (bf16*)alloc(262144 * 2);
  bf16* w_fc2_b = (bf16*)alloc(262144 * 2);
  bf16* w_q_b   = (bf16*)alloc(524288 * 2);
  bf16* w_k_b   = (bf16*)alloc(524288 * 2);
  bf16* w_v_b   = (bf16*)alloc(524288 * 2);
  bf16* w_pj_b  = (bf16*)alloc(524288 * 2);
  bf16* xc_b    = (bf16*)alloc((size_t)MTOK * 512 * 2);
  bf16* y_b     = (bf16*)alloc((size_t)MTOK * 512 * 2);
  bf16* mpre_b  = (bf16*)alloc((size_t)MTOK * 256 * 2);
  bf16* sa_b    = (bf16*)alloc((size_t)MTOK * 256 * 2);
  bf16* h1_b    = (bf16*)alloc((size_t)MTOK * 1024 * 2);   // reused by both branches
  bf16* Qb      = (bf16*)alloc((size_t)MTOK * 2048 * 2);   // reused as G after attention
  bf16* Kb      = (bf16*)alloc((size_t)MTOK * 2048 * 2);
  bf16* Vb      = (bf16*)alloc((size_t)MTOK * 2048 * 2);
  // f32 scratch
  float* xz_f    = (float*)alloc((size_t)MTOK * 1024 * 4);
  float* xcact_f = (float*)alloc((size_t)MTOK * 512 * 4);
  float* dbl_f   = (float*)alloc((size_t)MTOK * 48 * 4);
  float* dt_f    = (float*)alloc((size_t)MTOK * 512 * 4);
  float* ytmp_f  = (float*)alloc((size_t)MTOK * 512 * 4);

  dim3 blk(256);
  // weight conversions
  cvt_kernel<<<262144 / 256, blk, 0, stream>>>(in_proj_w, w_in_b, 262144);
  cvt_kernel<<<24576 / 256,  blk, 0, stream>>>(x_proj_w,  w_xp_b, 24576);
  cvt_kernel<<<131072 / 256, blk, 0, stream>>>(out_proj_w,w_out_b, 131072);
  cvt_kernel<<<262144 / 256, blk, 0, stream>>>(fc1_w,     w_fc1_b, 262144);
  cvt_kernel<<<262144 / 256, blk, 0, stream>>>(fc2_w,     w_fc2_b, 262144);
  cvt_kernel<<<524288 / 256, blk, 0, stream>>>(wq,        w_q_b, 524288);
  cvt_kernel<<<524288 / 256, blk, 0, stream>>>(wk,        w_k_b, 524288);
  cvt_kernel<<<524288 / 256, blk, 0, stream>>>(wv,        w_v_b, 524288);
  cvt_kernel<<<524288 / 256, blk, 0, stream>>>(proj_w,    w_pj_b, 524288);

  patchify_kernel<<<(MTOK * 256) / 256, blk, 0, stream>>>(x, q, xm_b, xsa_b, qp_b);

  // ---- Mamba branch ----
  gemm_bf16_wmma<0,0,0><<<dim3(128, 16), blk, 0, stream>>>(xm_b, w_in_b, nullptr, xz_f, nullptr, MTOK, 1024, 256, 0);
  conv_silu_kernel<<<(MTOK * 512) / 256, blk, 0, stream>>>(xz_f, conv_w, conv_b, xcact_f, xc_b);
  gemm_bf16_wmma<0,0,0><<<dim3(128, 1), blk, 0, stream>>>(xc_b, w_xp_b, nullptr, dbl_f, nullptr, MTOK, 48, 512, 0);
  dt_kernel<<<(MTOK * 512) / 256, blk, 0, stream>>>(dbl_f, dt_proj_w, dt_proj_b, dt_f);
  scan_kernel<<<(256 * 512) / 256, blk, 0, stream>>>(xcact_f, dt_f, dbl_f, xz_f, A_log, A_b_log, Dparam, ytmp_f, y_b);
  gemm_bf16_wmma<0,0,1><<<dim3(128, 4), blk, 0, stream>>>(y_b, w_out_b, nullptr, nullptr, mpre_b, MTOK, 256, 512, 0);
  gemm_bf16_wmma<0,2,1><<<dim3(128, 16), blk, 0, stream>>>(mpre_b, w_fc1_b, fc1_b, nullptr, h1_b, MTOK, 1024, 256, 0);
  gemm_bf16_wmma<0,1,3><<<dim3(128, 4), blk, 0, stream>>>(h1_b, w_fc2_b, fc2_b, out, nullptr, MTOK, 256, 1024, 0);

  // ---- attention branch ----
  gemm_bf16_wmma<0,0,1><<<dim3(128, 32), blk, 0, stream>>>(qp_b,  w_q_b, nullptr, nullptr, Qb, MTOK, 2048, 256, 0);
  gemm_bf16_wmma<0,0,1><<<dim3(128, 32), blk, 0, stream>>>(xsa_b, w_k_b, nullptr, nullptr, Kb, MTOK, 2048, 256, 0);
  gemm_bf16_wmma<0,0,1><<<dim3(128, 32), blk, 0, stream>>>(xsa_b, w_v_b, nullptr, nullptr, Vb, MTOK, 2048, 256, 0);
  attn_kernel<<<2048, blk, 0, stream>>>(Qb, Kb, Vb, Qb /* G aliases Q: disjoint per-block regions */);
  gemm_bf16_wmma<1,1,1><<<dim3(128, 4), blk, 0, stream>>>(Qb, w_pj_b, proj_b, nullptr, sa_b, MTOK, 256, 2048, 0);
  gemm_bf16_wmma<0,2,1><<<dim3(128, 16), blk, 0, stream>>>(sa_b, w_fc1_b, fc1_b, nullptr, h1_b, MTOK, 1024, 256, 0);
  gemm_bf16_wmma<0,1,3><<<dim3(128, 4), blk, 0, stream>>>(h1_b, w_fc2_b, fc2_b, out, nullptr, MTOK, 256, 1024, 256);
}
